// StableQGFDLayer_31104153157930
// MI455X (gfx1250) — matrix operand order repaired and checked by reading, not compiled, not run on hardware
//
#include <hip/hip_runtime.h>
#include <hip/hip_bf16.h>
#include <stdint.h>

typedef __attribute__((ext_vector_type(16))) __bf16 v16bf;
typedef __attribute__((ext_vector_type(8)))  float  v8f;

#define BDIM   2
#define SEQ    2048
#define DMODEL 1024
#define NHEAD  16
#define HDIM   64
#define ALPHA_ 0.05f

// ------------------------- helpers -------------------------
__device__ __forceinline__ unsigned short f2bf(float f) {
    unsigned int u = __float_as_uint(f);
    u += 0x7FFFu + ((u >> 16) & 1u);            // round-to-nearest-even
    return (unsigned short)(u >> 16);
}

union FragU {                // 32 bytes: 16 bf16 = one WMMA A/B fragment (wave32)
    uint4 q[2];
    v16bf v;
};

// ------------------------- WMMA GEMM (A * B^T form only) -------------------------
// C[M,N] = scale * A[M,K] * Bt^T (+bias[n]).
//   A  bf16 [M,K] row-major, leading dim lda
//   Bt bf16 [N,K] row-major, leading dim ldb   (i.e. C = A * Bt^T)
// Batched over blockIdx.z: zb = z/zH, zh = z%zH; operand offset = zb*s?1 + zh*s?2.
// Block tile 128x64, 256 threads = 8 waves (4x2), each wave 32x32 via 2x2 WMMA tiles.
// Double-buffered LDS: one __syncthreads per 32-wide K slab.
__global__ __launch_bounds__(256)
void gemm_bt_bf16_wmma(const unsigned short* __restrict__ A,
                       const unsigned short* __restrict__ Bt,
                       float* __restrict__ outF,
                       unsigned short* __restrict__ outB,
                       const float* __restrict__ bias,
                       float scale,
                       int K, int lda, int ldb, int ldc, int zH,
                       long long sA1, long long sA2,
                       long long sB1, long long sB2,
                       long long sC1, long long sC2)
{
    // row stride padded to 40 halves (80B, multiple of 16B) to avoid bank conflicts
    __shared__ __align__(16) unsigned short As[2][128 * 40];
    __shared__ __align__(16) unsigned short Bs[2][64 * 40];

    const int t  = threadIdx.x;
    const int bm = blockIdx.y * 128;
    const int bn = blockIdx.x * 64;
    const int zb = (int)blockIdx.z / zH;
    const int zh = (int)blockIdx.z % zH;

    const unsigned short* Ab = A  + zb * sA1 + zh * sA2;
    const unsigned short* Bb = Bt + zb * sB1 + zh * sB2;
    const long long coff = zb * sC1 + zh * sC2;

    const int wave = t >> 5;
    const int lane = t & 31;
    const int wm = (wave & 3) * 32;   // wave row offset inside block tile
    const int wn = (wave >> 2) * 32;  // wave col offset
    const int lm = lane & 15;
    const int hi = lane >> 4;

    // stagers: A tile 128x32 halves = 512 uint4 (2/thread); B tile 64x32 = 256 uint4 (1/thread)
    const int ar0 = t >> 2;               // 0..63 (+64 on second pass)
    const int ac8 = (t & 3) * 8;          // 0,8,16,24
    const int brn = t >> 2;               // 0..63
    const int bc8 = (t & 3) * 8;

    v8f acc[2][2] = {};

    // ---- prologue: stage slab k0=0 into buffer 0 ----
    {
        #pragma unroll
        for (int i = 0; i < 2; ++i) {
            const int r = ar0 + i * 64;
            const uint4 v = *reinterpret_cast<const uint4*>(Ab + (long long)(bm + r) * lda + ac8);
            *reinterpret_cast<uint4*>(&As[0][r * 40 + ac8]) = v;
        }
        const uint4 v = *reinterpret_cast<const uint4*>(Bb + (long long)(bn + brn) * ldb + bc8);
        *reinterpret_cast<uint4*>(&Bs[0][brn * 40 + bc8]) = v;
    }
    __syncthreads();

    int p = 0;
    for (int k0 = 0; k0 < K; k0 += 32) {
        const bool has_next = (k0 + 32) < K;

        // ---- issue global loads for next slab early ----
        uint4 na0, na1, nb0;
        if (has_next) {
            na0 = *reinterpret_cast<const uint4*>(Ab + (long long)(bm + ar0)      * lda + (k0 + 32 + ac8));
            na1 = *reinterpret_cast<const uint4*>(Ab + (long long)(bm + ar0 + 64) * lda + (k0 + 32 + ac8));
            nb0 = *reinterpret_cast<const uint4*>(Bb + (long long)(bn + brn)      * ldb + (k0 + 32 + bc8));
            if (k0 + 64 < K)   // prefetch slab after next (gfx1250 global_prefetch_b8)
                __builtin_prefetch(Ab + (long long)(bm + ar0) * lda + (k0 + 64 + ac8), 0, 1);
        }

        // ---- fragments from buffer p ----
        FragU a[2], b[2];
        #pragma unroll
        for (int i = 0; i < 2; ++i) {
            const int row = wm + i * 16 + lm;
            a[i].q[0] = *reinterpret_cast<const uint4*>(&As[p][row * 40 + hi * 8]);
            a[i].q[1] = *reinterpret_cast<const uint4*>(&As[p][row * 40 + 16 + hi * 8]);
        }
        #pragma unroll
        for (int j = 0; j < 2; ++j) {
            const int col = wn + j * 16 + lm;
            b[j].q[0] = *reinterpret_cast<const uint4*>(&Bs[p][col * 40 + hi * 8]);
            b[j].q[1] = *reinterpret_cast<const uint4*>(&Bs[p][col * 40 + 16 + hi * 8]);
        }

        // ---- 4x WMMA ----
        #pragma unroll
        for (int i = 0; i < 2; ++i)
            #pragma unroll
            for (int j = 0; j < 2; ++j)
                acc[i][j] = __builtin_amdgcn_wmma_f32_16x16x32_bf16(
                    false, a[i].v, false, b[j].v, (short)0, acc[i][j], false, false);

        // ---- commit next slab into the other buffer; single barrier per slab ----
        if (has_next) {
            const int q = p ^ 1;
            *reinterpret_cast<uint4*>(&As[q][ ar0       * 40 + ac8]) = na0;
            *reinterpret_cast<uint4*>(&As[q][(ar0 + 64) * 40 + ac8]) = na1;
            *reinterpret_cast<uint4*>(&Bs[q][ brn       * 40 + bc8]) = nb0;
            __syncthreads();
            p = q;
        }
    }

    // ---- epilogue: C/D layout -> row = hi*8 + v, col = lane%16 ----
    #pragma unroll
    for (int j = 0; j < 2; ++j) {
        const int col = bn + wn + j * 16 + lm;
        const float bi = bias ? bias[col] : 0.0f;
        #pragma unroll
        for (int i = 0; i < 2; ++i) {
            #pragma unroll
            for (int v = 0; v < 8; ++v) {
                const int row = bm + wm + i * 16 + hi * 8 + v;
                const float val = acc[i][j][v] * scale + bi;
                const long long idx = coff + (long long)row * ldc + col;
                if (outF) outF[idx] = val;
                if (outB) outB[idx] = f2bf(val);
            }
        }
    }
}

// ------------------------- row softmax (in place, f32) -------------------------
__global__ __launch_bounds__(256)
void softmax_rows(float* __restrict__ buf, int n)
{
    __shared__ float red[256];
    float* row = buf + (long long)blockIdx.x * n;
    const int t = threadIdx.x;

    float mx = -3.4e38f;
    for (int i = t; i < n; i += 256) mx = fmaxf(mx, row[i]);
    red[t] = mx; __syncthreads();
    for (int s = 128; s > 0; s >>= 1) { if (t < s) red[t] = fmaxf(red[t], red[t + s]); __syncthreads(); }
    mx = red[0]; __syncthreads();

    float sum = 0.0f;
    for (int i = t; i < n; i += 256) { float e = __expf(row[i] - mx); row[i] = e; sum += e; }
    red[t] = sum; __syncthreads();
    for (int s = 128; s > 0; s >>= 1) { if (t < s) red[t] += red[t + s]; __syncthreads(); }
    const float inv = 1.0f / red[0];
    for (int i = t; i < n; i += 256) row[i] *= inv;
}

// ------------------------- elementwise -------------------------
__global__ void cvt_f32_bf16(const float* __restrict__ in,
                             unsigned short* __restrict__ out, long long count)
{
    long long i = (long long)blockIdx.x * blockDim.x + threadIdx.x;
    const long long stride = (long long)gridDim.x * blockDim.x;
    for (; i < count; i += stride) out[i] = f2bf(in[i]);
}

__global__ void blend_kernel(const float* __restrict__ p0, const float* __restrict__ tt,
                             float* __restrict__ outF, unsigned short* __restrict__ outB,
                             long long count)
{
    long long i = (long long)blockIdx.x * blockDim.x + threadIdx.x;
    const long long stride = (long long)gridDim.x * blockDim.x;
    for (; i < count; i += stride) {
        const float v = (1.0f - ALPHA_) * p0[i] + ALPHA_ * tt[i];
        if (outF) outF[i] = v;
        if (outB) outB[i] = f2bf(v);
    }
}

// ------------------------- tiled transposes (32x32 tiles, 256 threads) -------------------------
// out[c][r] = cvt(in[r][c]); batched via blockIdx.z with strides sIn/sOut (elements)
__global__ __launch_bounds__(256)
void transpose_f32_bf16(const float* __restrict__ in, unsigned short* __restrict__ out,
                        int rows, int cols, long long sIn, long long sOut)
{
    __shared__ float tile[32][33];
    const long long ib = (long long)blockIdx.z * sIn;
    const long long ob = (long long)blockIdx.z * sOut;
    const int r0 = blockIdx.y * 32, c0 = blockIdx.x * 32;
    const int tx = threadIdx.x & 31, ty = threadIdx.x >> 5;   // 32 x 8
    #pragma unroll
    for (int i = 0; i < 32; i += 8)
        tile[ty + i][tx] = in[ib + (long long)(r0 + ty + i) * cols + (c0 + tx)];
    __syncthreads();
    #pragma unroll
    for (int i = 0; i < 32; i += 8)
        out[ob + (long long)(c0 + ty + i) * rows + (r0 + tx)] = f2bf(tile[tx][ty + i]);
}

__global__ __launch_bounds__(256)
void transpose_bf16(const unsigned short* __restrict__ in, unsigned short* __restrict__ out,
                    int rows, int cols, long long sIn, long long sOut)
{
    __shared__ unsigned short tile[32][34];
    const long long ib = (long long)blockIdx.z * sIn;
    const long long ob = (long long)blockIdx.z * sOut;
    const int r0 = blockIdx.y * 32, c0 = blockIdx.x * 32;
    const int tx = threadIdx.x & 31, ty = threadIdx.x >> 5;
    #pragma unroll
    for (int i = 0; i < 32; i += 8)
        tile[ty + i][tx] = in[ib + (long long)(r0 + ty + i) * cols + (c0 + tx)];
    __syncthreads();
    #pragma unroll
    for (int i = 0; i < 32; i += 8)
        out[ob + (long long)(c0 + ty + i) * rows + (r0 + tx)] = tile[tx][ty + i];
}

// ------------------------- launcher -------------------------
extern "C" void kernel_launch(void* const* d_in, const int* in_sizes, int n_in,
                              void* d_out, int out_size, void* d_ws, size_t ws_size,
                              hipStream_t stream)
{
    const float* X  = (const float*)d_in[0];
    const float* wq = (const float*)d_in[1];
    const float* bq = (const float*)d_in[2];
    const float* wk = (const float*)d_in[3];
    const float* bk = (const float*)d_in[4];
    const float* wv = (const float*)d_in[5];
    const float* bv = (const float*)d_in[6];
    const float* wo = (const float*)d_in[7];
    const float* bo = (const float*)d_in[8];

    float* out_proj = (float*)d_out;                                    // [B,N,D]
    float* p_out    = out_proj + (long long)BDIM * SEQ * DMODEL;        // [B,H,N,N]

    const long long MROWS = (long long)BDIM * SEQ;                      // 4096
    const long long PC    = (long long)BDIM * NHEAD * SEQ * SEQ;        // 134M

    // workspace carve-out
    char* ws = (char*)d_ws;
    size_t off = 0;
    auto carve = [&](size_t bytes) -> void* {
        void* p = ws + off;
        off += (bytes + 255) & ~(size_t)255;
        return p;
    };
    unsigned short* Xb    = (unsigned short*)carve(MROWS * DMODEL * 2);
    unsigned short* wqT   = (unsigned short*)carve((size_t)DMODEL * DMODEL * 2);
    unsigned short* wkT   = (unsigned short*)carve((size_t)DMODEL * DMODEL * 2);
    unsigned short* wvT   = (unsigned short*)carve((size_t)DMODEL * DMODEL * 2);
    unsigned short* woT   = (unsigned short*)carve((size_t)DMODEL * DMODEL * 2);
    unsigned short* Qb    = (unsigned short*)carve(MROWS * DMODEL * 2);
    unsigned short* Kb    = (unsigned short*)carve(MROWS * DMODEL * 2);
    unsigned short* Vb    = (unsigned short*)carve(MROWS * DMODEL * 2);
    unsigned short* VbT   = (unsigned short*)carve(MROWS * DMODEL * 2);   // [B][D][SEQ]
    unsigned short* attnb = (unsigned short*)carve(MROWS * DMODEL * 2);
    float*          p0f   = (float*)carve((size_t)PC * 4);
    float*          S2    = (float*)carve((size_t)PC * 4);   // sim -> P -> diffusion temp
    unsigned short* p0b   = (unsigned short*)carve((size_t)PC * 2); // p0 -> p1 -> p (bf16)
    unsigned short* PbT   = (unsigned short*)carve((size_t)PC * 2); // P^T per (b,h)

    const dim3 blk(256);

    // 1) activations to bf16; weights to transposed bf16 (W^T so all GEMMs are A*Bt^T)
    cvt_f32_bf16<<<2048, 256, 0, stream>>>(X, Xb, MROWS * DMODEL);
    const dim3 gwt(DMODEL / 32, DMODEL / 32, 1);
    transpose_f32_bf16<<<gwt, blk, 0, stream>>>(wq, wqT, DMODEL, DMODEL, 0, 0);
    transpose_f32_bf16<<<gwt, blk, 0, stream>>>(wk, wkT, DMODEL, DMODEL, 0, 0);
    transpose_f32_bf16<<<gwt, blk, 0, stream>>>(wv, wvT, DMODEL, DMODEL, 0, 0);
    transpose_f32_bf16<<<gwt, blk, 0, stream>>>(wo, woT, DMODEL, DMODEL, 0, 0);

    // 2) QKV projections: [4096,1024] = Xb @ W + b  (bf16 out)
    const dim3 gproj(DMODEL / 64, (unsigned)(MROWS / 128), 1);
    gemm_bt_bf16_wmma<<<gproj, blk, 0, stream>>>(Xb, wqT, nullptr, Qb, bq, 1.0f,
        DMODEL, DMODEL, DMODEL, DMODEL, 1, 0,0, 0,0, 0,0);
    gemm_bt_bf16_wmma<<<gproj, blk, 0, stream>>>(Xb, wkT, nullptr, Kb, bk, 1.0f,
        DMODEL, DMODEL, DMODEL, DMODEL, 1, 0,0, 0,0, 0,0);
    gemm_bt_bf16_wmma<<<gproj, blk, 0, stream>>>(Xb, wvT, nullptr, Vb, bv, 1.0f,
        DMODEL, DMODEL, DMODEL, DMODEL, 1, 0,0, 0,0, 0,0);

    // 3) scores = Q K^T / sqrt(HD) per (b,h); softmax in place -> p0 (f32)
    const dim3 ghead(SEQ / 64, SEQ / 128, BDIM * NHEAD);
    gemm_bt_bf16_wmma<<<ghead, blk, 0, stream>>>(Qb, Kb, p0f, nullptr, nullptr, 0.125f,
        HDIM, DMODEL, DMODEL, SEQ, NHEAD,
        (long long)SEQ * DMODEL, 64,
        (long long)SEQ * DMODEL, 64,
        (long long)NHEAD * SEQ * SEQ, (long long)SEQ * SEQ);
    softmax_rows<<<BDIM * NHEAD * SEQ, 256, 0, stream>>>(p0f, SEQ);

    // 4) sim = K K^T / sqrt(D); softmax -> P (f32 in S2); transpose to bf16 P^T
    gemm_bt_bf16_wmma<<<ghead, blk, 0, stream>>>(Kb, Kb, S2, nullptr, nullptr, 0.03125f,
        HDIM, DMODEL, DMODEL, SEQ, NHEAD,
        (long long)SEQ * DMODEL, 64,
        (long long)SEQ * DMODEL, 64,
        (long long)NHEAD * SEQ * SEQ, (long long)SEQ * SEQ);
    softmax_rows<<<BDIM * NHEAD * SEQ, 256, 0, stream>>>(S2, SEQ);
    const dim3 gpt(SEQ / 32, SEQ / 32, BDIM * NHEAD);
    transpose_f32_bf16<<<gpt, blk, 0, stream>>>(S2, PbT, SEQ, SEQ,
        (long long)SEQ * SEQ, (long long)SEQ * SEQ);
    cvt_f32_bf16<<<8192, 256, 0, stream>>>(p0f, p0b, PC);

    // 5) diffusion step 1: t1 = p0b @ P (= p0b * PbT^T); p1 = (1-a)p0 + a*t1 -> p0b
    gemm_bt_bf16_wmma<<<ghead, blk, 0, stream>>>(p0b, PbT, S2, nullptr, nullptr, 1.0f,
        SEQ, SEQ, SEQ, SEQ, 1,
        (long long)SEQ * SEQ, 0, (long long)SEQ * SEQ, 0, (long long)SEQ * SEQ, 0);
    blend_kernel<<<8192, 256, 0, stream>>>(p0f, S2, nullptr, p0b, PC);

    // 6) diffusion step 2: t2 = p1b @ P; p = (1-a)p0 + a*t2 -> d_out (f32) + p0b (bf16)
    gemm_bt_bf16_wmma<<<ghead, blk, 0, stream>>>(p0b, PbT, S2, nullptr, nullptr, 1.0f,
        SEQ, SEQ, SEQ, SEQ, 1,
        (long long)SEQ * SEQ, 0, (long long)SEQ * SEQ, 0, (long long)SEQ * SEQ, 0);
    blend_kernel<<<8192, 256, 0, stream>>>(p0f, S2, p_out, p0b, PC);

    // 7) V^T per batch: VbT[b][c][n] = V[b][n][c]
    const dim3 gvt(DMODEL / 32, SEQ / 32, BDIM);
    transpose_bf16<<<gvt, blk, 0, stream>>>(Vb, VbT, SEQ, DMODEL,
        (long long)SEQ * DMODEL, (long long)DMODEL * SEQ);

    // 8) attn = p @ V per head -> attnb bf16 with heads merged (ldc = D)
    const dim3 gattn(HDIM / 64, SEQ / 128, BDIM * NHEAD);
    gemm_bt_bf16_wmma<<<gattn, blk, 0, stream>>>(p0b, VbT, nullptr, attnb, nullptr, 1.0f,
        SEQ, SEQ, SEQ, DMODEL, NHEAD,
        (long long)NHEAD * SEQ * SEQ, (long long)SEQ * SEQ,
        (long long)DMODEL * SEQ, (long long)HDIM * SEQ,
        (long long)SEQ * DMODEL, 64);

    // 9) out = attn @ wo + bo -> d_out (f32)
    gemm_bt_bf16_wmma<<<gproj, blk, 0, stream>>>(attnb, woT, out_proj, nullptr, bo, 1.0f,
        DMODEL, DMODEL, DMODEL, DMODEL, 1, 0,0, 0,0, 0,0);
}
